// KgAdapterRGAT_71442486002190
// MI455X (gfx1250) — matrix operand order repaired and checked by reading, not compiled, not run on hardware
//
#include <hip/hip_runtime.h>

typedef _Float16 f16;
typedef _Float16 v16h __attribute__((ext_vector_type(16)));
typedef _Float16 v8h  __attribute__((ext_vector_type(8)));
typedef _Float16 v4h  __attribute__((ext_vector_type(4)));
typedef float    v8f  __attribute__((ext_vector_type(8)));

#define D    128
#define RREL 8

// ---------- order-preserving float <-> uint for atomic max ----------
__device__ __forceinline__ unsigned enc_f(float f) {
  unsigned u = __float_as_uint(f);
  return (u & 0x80000000u) ? ~u : (u | 0x80000000u);
}
__device__ __forceinline__ float dec_f(unsigned v) {
  unsigned u = (v & 0x80000000u) ? (v & 0x7fffffffu) : ~v;
  return __uint_as_float(u);
}

// ---------- small utility kernels ----------
__global__ void k_f32_to_f16(const float* __restrict__ x, f16* __restrict__ o, int n) {
  int i = blockIdx.x * blockDim.x + threadIdx.x;
  if (i < n) o[i] = (f16)x[i];
}

__global__ void k_zero32(unsigned* __restrict__ p, int n) {
  int i = blockIdx.x * blockDim.x + threadIdx.x;
  if (i < n) p[i] = 0u;   // 0u == encoded -max for amax, == 0.0f for floats
}

__global__ void k_copy_i64(const long long* __restrict__ s, long long* __restrict__ d, int n) {
  int i = blockIdx.x * blockDim.x + threadIdx.x;
  if (i < n) d[i] = s[i];
}

// ---------- pack weights into WMMA B-fragment layout ----------
// B (32x16 f16, K x N): lane half h, element e<8 -> K=8h+e, e>=8 -> K=16+8h+(e-8)
// (matches A fragment order).  Packed layout: [r][ot(8)][kb(4)][lane(32)][e(16)]
__global__ void k_pack_b(const float* __restrict__ w, f16* __restrict__ bp,
                         int nrel, int transpose) {
  int t = blockIdx.x * blockDim.x + threadIdx.x;
  if (t >= nrel * 1024) return;
  int lane = t & 31;
  int kb   = (t >> 5) & 3;
  int ot   = (t >> 7) & 7;
  int r    = t >> 10;
  int m = lane & 15, hsel = lane >> 4;
  int col = ot * 16 + m;
  const float* wr = w + (size_t)r * D * D;
  f16* dst = bp + (size_t)t * 16;
#pragma unroll
  for (int e = 0; e < 16; ++e) {
    int k = 32 * kb + (e < 8 ? 8 * hsel + e : 16 + 8 * hsel + (e - 8));
    float v = transpose ? wr[col * D + k] : wr[k * D + col];
    dst[e] = (f16)v;
  }
}

// wq[r][i] = sum_o w[r][i][o]*q[r][o] ;  wk likewise
__global__ void k_wqk(const float* __restrict__ w, const float* __restrict__ q,
                      const float* __restrict__ k, float* __restrict__ wq,
                      float* __restrict__ wk) {
  int t = blockIdx.x * blockDim.x + threadIdx.x;
  if (t >= RREL * D) return;
  int r = t >> 7, i = t & (D - 1);
  const float* wrow = w + ((size_t)r * D + i) * D;
  const float* qr = q + r * D;
  const float* kr = k + r * D;
  float sq = 0.f, sk = 0.f;
  for (int o = 0; o < D; ++o) { float v = wrow[o]; sq += v * qr[o]; sk += v * kr[o]; }
  wq[t] = sq; wk[t] = sk;
}

// xq[n][r] = x[n,:] . wq[r,:]  (fp32), wq/wk cached in LDS
__global__ void k_xqk(const float* __restrict__ x, const float* __restrict__ wq,
                      const float* __restrict__ wk, float* __restrict__ xq,
                      float* __restrict__ xk, int n_nodes) {
  __shared__ float swq[RREL * D];
  __shared__ float swk[RREL * D];
  for (int i = threadIdx.x; i < RREL * D; i += blockDim.x) { swq[i] = wq[i]; swk[i] = wk[i]; }
  __syncthreads();
  int t = blockIdx.x * blockDim.x + threadIdx.x;
  if (t >= n_nodes * RREL) return;
  int n = t >> 3, r = t & 7;
  const float* xr = x + (size_t)n * D;
  const float* qv = swq + r * D;
  const float* kv = swk + r * D;
  float sq = 0.f, sk = 0.f;
  for (int i = 0; i < D; ++i) { float v = xr[i]; sq += v * qv[i]; sk += v * kv[i]; }
  xq[t] = sq; xk[t] = sk;
}

// ---------- WMMA GEMM ----------
// A fragment (16x32 f16): lane = m + 16*hsel; elements 0..7 = K 8*hsel..8*hsel+7,
// elements 8..15 = K 16+8*hsel..16+8*hsel+7 -> two contiguous 16B loads.
__device__ __forceinline__ v16h load_a_frag(const f16* __restrict__ arow, int kb, int hsel) {
  const v8h lo = *(const v8h*)(arow + 32 * kb + 8 * hsel);
  const v8h hi = *(const v8h*)(arow + 32 * kb + 16 + 8 * hsel);
  return __builtin_shufflevector(lo, hi, 0,1,2,3,4,5,6,7,8,9,10,11,12,13,14,15);
}

// xw[n][r][o] = x[n,:] @ w[r]; one block = 8 waves = 8 col-tiles of one (16-row, r) tile
__global__ void __launch_bounds__(256) k_gemm_rel(const f16* __restrict__ A,
                                                  const f16* __restrict__ Bp,
                                                  f16* __restrict__ xwh) {
  int bid = blockIdx.x;
  int r  = bid & 7;
  int nt = bid >> 3;
  int tid = threadIdx.x;
  int wave = tid >> 5;
  int lane = tid & 31;
  int m = lane & 15, hsel = lane >> 4;
  int row0 = nt * 16;
  const f16* arow  = A + (size_t)(row0 + m) * D;
  const f16* bfrag = Bp + ((size_t)(r * 8 + wave) * 4 * 32) * 16 + lane * 16;
  v8f c = {};
#pragma unroll
  for (int kb = 0; kb < 4; ++kb) {
    v16h a = load_a_frag(arow, kb, hsel);
    v16h b = *(const v16h*)(bfrag + (size_t)kb * 512);
    c = __builtin_amdgcn_wmma_f32_16x16x32_f16(false, a, false, b, (short)0, c, false, false);
  }
  // C layout: vgpr v -> row M=v+8*hsel, col N=lane&15
  f16* out = xwh + (((size_t)(row0 + 8 * hsel)) * RREL + r) * D + wave * 16 + m;
#pragma unroll
  for (int v = 0; v < 8; ++v) out[(size_t)v * RREL * D] = (f16)c[v];
}

// out[n][o] = h[n,:] @ lin_w.T + lin_b  (fp32 out)
__global__ void __launch_bounds__(256) k_gemm_lin(const f16* __restrict__ A,
                                                  const f16* __restrict__ Bp,
                                                  const float* __restrict__ bias,
                                                  float* __restrict__ out) {
  int nt = blockIdx.x;
  int tid = threadIdx.x;
  int wave = tid >> 5;
  int lane = tid & 31;
  int m = lane & 15, hsel = lane >> 4;
  int row0 = nt * 16;
  const f16* arow  = A + (size_t)(row0 + m) * D;
  const f16* bfrag = Bp + ((size_t)wave * 4 * 32) * 16 + lane * 16;
  v8f c = {};
#pragma unroll
  for (int kb = 0; kb < 4; ++kb) {
    v16h a = load_a_frag(arow, kb, hsel);
    v16h b = *(const v16h*)(bfrag + (size_t)kb * 512);
    c = __builtin_amdgcn_wmma_f32_16x16x32_f16(false, a, false, b, (short)0, c, false, false);
  }
  int col = wave * 16 + m;
  float bv = bias[col];
  float* op = out + (size_t)(row0 + 8 * hsel) * D + col;
#pragma unroll
  for (int v = 0; v < 8; ++v) op[(size_t)v * D] = c[v] + bv;
}

// ---------- attention edge kernels ----------
__global__ void k_edge_phase1(const long long* __restrict__ ei, const long long* __restrict__ et,
                              const float* __restrict__ xq, const float* __restrict__ xk,
                              float* __restrict__ raw, unsigned* __restrict__ amax, int E) {
  int e = blockIdx.x * blockDim.x + threadIdx.x;
  if (e >= E) return;
  int s = (int)ei[e], d = (int)ei[(size_t)E + e], t = (int)et[e];
  float a = xq[d * RREL + t] + xk[s * RREL + t];
  a = (a > 0.f) ? a : 0.2f * a;            // leaky relu, slope 0.2
  raw[e] = a;
  atomicMax(&amax[d], enc_f(a));
}

__global__ void k_edge_phase2(const long long* __restrict__ ei, float* __restrict__ raw,
                              const unsigned* __restrict__ amax, float* __restrict__ denom, int E) {
  int e = blockIdx.x * blockDim.x + threadIdx.x;
  if (e >= E) return;
  int d = (int)ei[(size_t)E + e];
  float ex = __expf(raw[e] - dec_f(amax[d]));
  raw[e] = ex;
  atomicAdd(&denom[d], ex);
}

__global__ void k_edge_phase3(const long long* __restrict__ ei, float* __restrict__ raw,
                              const float* __restrict__ denom, float* __restrict__ aout, int E) {
  int e = blockIdx.x * blockDim.x + threadIdx.x;
  if (e >= E) return;
  int d = (int)ei[(size_t)E + e];
  float al = raw[e] / (denom[d] + 1e-16f);
  raw[e] = al;
  if (aout) aout[e] = al;
}

// one wave per edge; each lane handles 4 channels (f16x4 load, 4 f32 atomics)
__global__ void k_scatter(const long long* __restrict__ ei, const long long* __restrict__ et,
                          const f16* __restrict__ xwh, const float* __restrict__ alpha,
                          float* __restrict__ h, int E) {
  int g = blockIdx.x * blockDim.x + threadIdx.x;
  int e = g >> 5;
  if (e >= E) return;
  int lane = g & 31;
  int s = (int)ei[e], d = (int)ei[(size_t)E + e], t = (int)et[e];
  float a = alpha[e];
  v4h mv = *(const v4h*)(xwh + ((size_t)s * RREL + t) * D + lane * 4);
  float* dp = h + (size_t)d * D + lane * 4;
  atomicAdd(dp + 0, a * (float)mv[0]);
  atomicAdd(dp + 1, a * (float)mv[1]);
  atomicAdd(dp + 2, a * (float)mv[2]);
  atomicAdd(dp + 3, a * (float)mv[3]);
}

__global__ void k_bias_relu(float* __restrict__ h, const float* __restrict__ b,
                            f16* __restrict__ act, int n) {
  int i = blockIdx.x * blockDim.x + threadIdx.x;
  if (i >= n) return;
  float v = h[i] + b[i & (D - 1)];
  v = v > 0.f ? v : 0.f;
  h[i] = v;
  act[i] = (f16)v;
}

// ---------- launch ----------
extern "C" void kernel_launch(void* const* d_in, const int* in_sizes, int n_in,
                              void* d_out, int out_size, void* d_ws, size_t ws_size,
                              hipStream_t stream) {
  (void)n_in; (void)out_size; (void)ws_size;
  const float*     x  = (const float*)d_in[0];
  const long long* ei = (const long long*)d_in[1];   // [2,E] int64
  const long long* et = (const long long*)d_in[2];   // [E]   int64
  const float* w1 = (const float*)d_in[3];
  const float* q1 = (const float*)d_in[4];
  const float* k1 = (const float*)d_in[5];
  const float* b1 = (const float*)d_in[6];
  const float* w2 = (const float*)d_in[7];
  const float* q2 = (const float*)d_in[8];
  const float* k2 = (const float*)d_in[9];
  const float* b2 = (const float*)d_in[10];
  const float* lw = (const float*)d_in[11];
  const float* lb = (const float*)d_in[12];

  const int NND = in_sizes[0];     // N*D
  const int NN  = NND / D;         // nodes
  const int E   = in_sizes[2];     // edges
  const int ntiles = NN / 16;

  // carve workspace
  char* ws = (char*)d_ws;
  size_t off = 0;
  auto carve = [&](size_t bytes) -> void* {
    void* p = ws + off;
    off += (bytes + 255) & ~(size_t)255;
    return p;
  };
  f16*   act16 = (f16*)carve((size_t)NND * 2);
  f16*   xwh   = (f16*)carve((size_t)NN * RREL * D * 2);
  float* h     = (float*)carve((size_t)NND * 4);
  f16*   w1p   = (f16*)carve((size_t)RREL * 1024 * 16 * 2);
  f16*   w2p   = (f16*)carve((size_t)RREL * 1024 * 16 * 2);
  f16*   lwp   = (f16*)carve((size_t)1024 * 16 * 2);
  float* wq1   = (float*)carve((size_t)RREL * D * 4);
  float* wk1   = (float*)carve((size_t)RREL * D * 4);
  float* wq2   = (float*)carve((size_t)RREL * D * 4);
  float* wk2   = (float*)carve((size_t)RREL * D * 4);
  float* xq    = (float*)carve((size_t)NN * RREL * 4);
  float* xk    = (float*)carve((size_t)NN * RREL * 4);
  float* ebuf  = (float*)carve((size_t)E * 4);
  unsigned* amax = (unsigned*)carve((size_t)NN * 4);
  float* denom   = (float*)carve((size_t)NN * 4);

  // output tuple layout: out f32 [N,D] | edge_index int64 [2,E] | alpha2 f32 [E,1]
  float*     out_x  = (float*)d_out;
  long long* out_ei = (long long*)((char*)d_out + (size_t)NND * 4);
  float*     out_al = (float*)((char*)d_out + (size_t)NND * 4 + (size_t)E * 2 * 8);

  const int B = 256;
  auto gd = [&](int n) { return (n + B - 1) / B; };

  // prep: activation f16, weight packs, fused q/k projections
  k_f32_to_f16<<<gd(NND), B, 0, stream>>>(x, act16, NND);
  k_pack_b<<<gd(RREL * 1024), B, 0, stream>>>(w1, w1p, RREL, 0);
  k_pack_b<<<gd(RREL * 1024), B, 0, stream>>>(w2, w2p, RREL, 0);
  k_pack_b<<<gd(1024), B, 0, stream>>>(lw, lwp, 1, 1);
  k_wqk<<<gd(RREL * D), B, 0, stream>>>(w1, q1, k1, wq1, wk1);
  k_wqk<<<gd(RREL * D), B, 0, stream>>>(w2, q2, k2, wq2, wk2);

  // ---- layer 1 ----
  k_gemm_rel<<<ntiles * RREL, B, 0, stream>>>(act16, w1p, xwh);
  k_xqk<<<gd(NN * RREL), B, 0, stream>>>(x, wq1, wk1, xq, xk, NN);
  k_zero32<<<gd(NND), B, 0, stream>>>((unsigned*)h, NND);
  k_zero32<<<gd(NN), B, 0, stream>>>(amax, NN);
  k_zero32<<<gd(NN), B, 0, stream>>>((unsigned*)denom, NN);
  k_edge_phase1<<<gd(E), B, 0, stream>>>(ei, et, xq, xk, ebuf, amax, E);
  k_edge_phase2<<<gd(E), B, 0, stream>>>(ei, ebuf, amax, denom, E);
  k_edge_phase3<<<gd(E), B, 0, stream>>>(ei, ebuf, denom, (float*)nullptr, E);
  k_scatter<<<gd(E * 32), B, 0, stream>>>(ei, et, xwh, ebuf, h, E);
  k_bias_relu<<<gd(NND), B, 0, stream>>>(h, b1, act16, NND);

  // ---- layer 2 ----
  k_gemm_rel<<<ntiles * RREL, B, 0, stream>>>(act16, w2p, xwh);
  k_xqk<<<gd(NN * RREL), B, 0, stream>>>(h, wq2, wk2, xq, xk, NN);   // before h reset
  k_zero32<<<gd(NND), B, 0, stream>>>((unsigned*)h, NND);
  k_zero32<<<gd(NN), B, 0, stream>>>(amax, NN);
  k_zero32<<<gd(NN), B, 0, stream>>>((unsigned*)denom, NN);
  k_edge_phase1<<<gd(E), B, 0, stream>>>(ei, et, xq, xk, ebuf, amax, E);
  k_edge_phase2<<<gd(E), B, 0, stream>>>(ei, ebuf, amax, denom, E);
  k_edge_phase3<<<gd(E), B, 0, stream>>>(ei, ebuf, denom, out_al, E);   // alpha2 out
  k_scatter<<<gd(E * 32), B, 0, stream>>>(ei, et, xwh, ebuf, h, E);
  k_bias_relu<<<gd(NND), B, 0, stream>>>(h, b2, act16, NND);

  // ---- final linear + passthrough outputs ----
  k_gemm_lin<<<ntiles, B, 0, stream>>>(act16, lwp, lb, out_x);
  k_copy_i64<<<gd(2 * E), B, 0, stream>>>(ei, out_ei, 2 * E);
}
